// PA_MIL_85512798863750
// MI455X (gfx1250) — compile-verified, matrix operand order checked by reading, not run on hardware
//
#include <hip/hip_runtime.h>
#include <hip/hip_bf16.h>

typedef __attribute__((ext_vector_type(16))) _Float16 v16h;
typedef __attribute__((ext_vector_type(8)))  _Float16 v8h;
typedef __attribute__((ext_vector_type(8)))  float    v8f;

#define EMBED    512
#define IN_DIM   1024
#define NROWS    8192   // 1 cls + 8191 patches
#define KCHUNK   512

// ---------------------------------------------------------------------------
// Conversion kernels
// ---------------------------------------------------------------------------

// X16: row 0 = zeros (cls placeholder), rows 1..8191 = x rows 0..8190, f16
__global__ void cvt_x_kernel(const float* __restrict__ x, _Float16* __restrict__ X16,
                             int total, int rowlen) {
    int idx = blockIdx.x * blockDim.x + threadIdx.x;
    if (idx < total) {
        X16[idx] = (idx < rowlen) ? (_Float16)0.0f : (_Float16)x[idx - rowlen];
    }
}

// dst[n*K + k] = (f16) src[k*N + n]   (store weights transposed, N x K row-major)
__global__ void transpose_cvt_kernel(const float* __restrict__ src, _Float16* __restrict__ dst,
                                     int K, int N) {
    int idx = blockIdx.x * blockDim.x + threadIdx.x;
    if (idx < K * N) {
        int n = idx / K;
        int k = idx - n * K;
        dst[idx] = (_Float16)src[k * N + n];
    }
}

// ---------------------------------------------------------------------------
// Generic WMMA GEMM:  C[M x N] = act( A[M x K] @ Bt[N x K]^T + bias )
// A, Bt row-major f16.  One wave -> one 16x16 C tile.  8 waves / block.
// ---------------------------------------------------------------------------
__global__ void wmma_gemm_kernel(const _Float16* __restrict__ A,
                                 const _Float16* __restrict__ Bt,
                                 const float* __restrict__ bias,
                                 float* __restrict__ C32,
                                 _Float16* __restrict__ C16,
                                 int K, int ldc, int tilesM, int tilesN, int relu_flag) {
    int wave = threadIdx.x >> 5;
    int lane = threadIdx.x & 31;
    int tile = blockIdx.x * (blockDim.x >> 5) + wave;
    if (tile >= tilesM * tilesN) return;
    int tm = tile / tilesN;
    int tn = tile - tm * tilesN;
    int l  = lane & 15;     // row within A tile / col within B tile
    int hi = lane >> 4;     // half-wave selects K sub-blocks

    // Per ISA 16-bit A/B layout: this lane holds K = hi*8 + {0..7} (vec elems 0..7)
    // and K = hi*8 + 16 + {0..7} (vec elems 8..15) -> two contiguous 16B loads.
    const _Float16* arow = A  + (size_t)(tm * 16 + l) * K + hi * 8;
    const _Float16* brow = Bt + (size_t)(tn * 16 + l) * K + hi * 8;

    v8f acc = {};
    for (int k = 0; k < K; k += 32) {
        v8h a0 = *(const v8h*)(arow + k);
        v8h a1 = *(const v8h*)(arow + k + 16);
        v8h b0 = *(const v8h*)(brow + k);
        v8h b1 = *(const v8h*)(brow + k + 16);
        v16h av = __builtin_shufflevector(a0, a1, 0,1,2,3,4,5,6,7,8,9,10,11,12,13,14,15);
        v16h bv = __builtin_shufflevector(b0, b1, 0,1,2,3,4,5,6,7,8,9,10,11,12,13,14,15);
        acc = __builtin_amdgcn_wmma_f32_16x16x32_f16(false, av, false, bv,
                                                     (short)0, acc, false, false);
    }

    // C layout: lanes 0-15: VGPR i -> M=i ; lanes 16-31: VGPR i -> M=8+i ; N = lane&15
    int col = tn * 16 + l;
    float bv_ = bias ? bias[col] : 0.0f;
    #pragma unroll
    for (int i = 0; i < 8; ++i) {
        int row = tm * 16 + hi * 8 + i;
        float v = acc[i] + bv_;
        if (relu_flag) v = v > 0.0f ? v : 0.0f;
        if (C32) C32[(size_t)row * ldc + col] = v;
        if (C16) C16[(size_t)row * ldc + col] = (_Float16)v;
    }
}

// ---------------------------------------------------------------------------
// Row-0 (CLS) path kernels
// ---------------------------------------------------------------------------

__global__ void fix_row0_kernel(const float* __restrict__ cls,
                                float* __restrict__ H32, _Float16* __restrict__ Hh) {
    int t = threadIdx.x;            // 512 threads
    float c = cls[t];
    H32[t] = c;
    Hh[t]  = (_Float16)c;
}

// q0[n] = bq[n] + sum_k H32row0[k] * Wq[k*512 + n]
__global__ void q0_kernel(const float* __restrict__ H32row0, const float* __restrict__ Wq,
                          const float* __restrict__ bq, float* __restrict__ q0) {
    __shared__ float h0[EMBED];
    int t = threadIdx.x;            // 512 threads
    h0[t] = H32row0[t];
    __syncthreads();
    float acc = bq[t];
    for (int k = 0; k < EMBED; ++k) acc += h0[k] * Wq[k * EMBED + t];
    q0[t] = acc;
}

// s0[j] = (q0 . K16[j]) / sqrt(512)
__global__ void scores_row0_kernel(const float* __restrict__ q0,
                                   const _Float16* __restrict__ K16,
                                   float* __restrict__ s0) {
    __shared__ float q[EMBED];
    int t = threadIdx.x;            // 256 threads
    q[t]       = q0[t];
    q[t + 256] = q0[t + 256];
    __syncthreads();
    int j = blockIdx.x * 256 + t;
    const _Float16* krow = K16 + (size_t)j * EMBED;
    float acc = 0.0f;
    for (int k = 0; k < EMBED; ++k) acc += q[k] * (float)krow[k];
    s0[j] = acc * 0.04419417382415922f;   // 1/sqrt(512)
}

__device__ __forceinline__ float blk_reduce_max(float v, float* red, int t, int n) {
    red[t] = v; __syncthreads();
    for (int s = n >> 1; s > 0; s >>= 1) {
        if (t < s) red[t] = fmaxf(red[t], red[t + s]);
        __syncthreads();
    }
    float r = red[0]; __syncthreads();
    return r;
}
__device__ __forceinline__ float blk_reduce_sum(float v, float* red, int t, int n) {
    red[t] = v; __syncthreads();
    for (int s = n >> 1; s > 0; s >>= 1) {
        if (t < s) red[t] += red[t + s];
        __syncthreads();
    }
    float r = red[0]; __syncthreads();
    return r;
}

// Two-level softmax over row 0: softmax within each 512-chunk, then softmax over all 8192.
__global__ void softmax_row0_kernel(const float* __restrict__ s0, float* __restrict__ w0) {
    __shared__ float sm[NROWS];     // 32 KB
    __shared__ float red[512];
    int t = threadIdx.x;            // 512 threads
    for (int i = t; i < NROWS; i += 512) sm[i] = s0[i];
    __syncthreads();
    for (int c = 0; c < NROWS / KCHUNK; ++c) {
        float v = sm[c * KCHUNK + t];
        float m = blk_reduce_max(v, red, t, 512);
        float e = __expf(v - m);
        float s = blk_reduce_sum(e, red, t, 512);
        sm[c * KCHUNK + t] = e / s;
        __syncthreads();
    }
    float lmax = -1e30f;
    for (int i = t; i < NROWS; i += 512) lmax = fmaxf(lmax, sm[i]);
    float M = blk_reduce_max(lmax, red, t, 512);
    float lsum = 0.0f;
    for (int i = t; i < NROWS; i += 512) lsum += __expf(sm[i] - M);
    float S = blk_reduce_sum(lsum, red, t, 512);
    for (int i = t; i < NROWS; i += 512) w0[i] = __expf(sm[i] - M) / S;
}

// attn0[d] = sum_k w0[k] * V16[k*512 + d]
__global__ void attn0_kernel(const float* __restrict__ w0, const _Float16* __restrict__ V16,
                             float* __restrict__ attn0) {
    __shared__ float w[NROWS];      // 32 KB
    int t = threadIdx.x;            // 256 threads, 2 blocks
    for (int i = t; i < NROWS; i += 256) w[i] = w0[i];
    __syncthreads();
    int d = blockIdx.x * 256 + t;
    float acc = 0.0f;
    for (int k = 0; k < NROWS; ++k) acc += w[k] * (float)V16[(size_t)k * EMBED + d];
    attn0[d] = acc;
}

// h2 = H32row0 + attn0@Wo + bo ; LayerNorm ; logits = hn@Wc + bc
__global__ void final_kernel(const float* __restrict__ H32row0, const float* __restrict__ attn0,
                             const float* __restrict__ Wo, const float* __restrict__ bo,
                             const float* __restrict__ ln_g, const float* __restrict__ ln_b,
                             const float* __restrict__ Wc, const float* __restrict__ bc,
                             float* __restrict__ out) {
    __shared__ float a0[EMBED];
    __shared__ float red[512];
    int t = threadIdx.x;            // 512 threads
    a0[t] = attn0[t];
    __syncthreads();
    float proj = bo[t];
    for (int n = 0; n < EMBED; ++n) proj += a0[n] * Wo[n * EMBED + t];
    float h2 = H32row0[t] + proj;
    float mu  = blk_reduce_sum(h2, red, t, 512) * (1.0f / EMBED);
    float dd  = h2 - mu;
    float var = blk_reduce_sum(dd * dd, red, t, 512) * (1.0f / EMBED);
    float hn  = dd * rsqrtf(var + 1e-5f) * ln_g[t] + ln_b[t];
    float l0 = blk_reduce_sum(hn * Wc[t * 2 + 0], red, t, 512);
    float l1 = blk_reduce_sum(hn * Wc[t * 2 + 1], red, t, 512);
    if (t == 0) { out[0] = l0 + bc[0]; out[1] = l1 + bc[1]; }
}

// ---------------------------------------------------------------------------
// Launcher
// ---------------------------------------------------------------------------
extern "C" void kernel_launch(void* const* d_in, const int* in_sizes, int n_in,
                              void* d_out, int out_size, void* d_ws, size_t ws_size,
                              hipStream_t stream) {
    const float* x    = (const float*)d_in[0];
    const float* W1   = (const float*)d_in[1];
    const float* b1   = (const float*)d_in[2];
    const float* cls  = (const float*)d_in[3];
    const float* Wq   = (const float*)d_in[4];
    const float* bq   = (const float*)d_in[5];
    const float* Wk   = (const float*)d_in[6];
    const float* bk   = (const float*)d_in[7];
    const float* Wv   = (const float*)d_in[8];
    const float* bv   = (const float*)d_in[9];
    const float* Wo   = (const float*)d_in[10];
    const float* bo   = (const float*)d_in[11];
    const float* ln_g = (const float*)d_in[12];
    const float* ln_b = (const float*)d_in[13];
    const float* Wc   = (const float*)d_in[14];
    const float* bc   = (const float*)d_in[15];
    float* out = (float*)d_out;
    (void)in_sizes; (void)n_in; (void)out_size; (void)ws_size; (void)ln_g; (void)ln_b;

    char* ws = (char*)d_ws;
    size_t off = 0;
    auto wsalloc = [&](size_t bytes) -> void* {
        void* p = ws + off;
        off += (bytes + 255) & ~(size_t)255;
        return p;
    };
    _Float16* X16  = (_Float16*)wsalloc((size_t)NROWS * IN_DIM * 2);   // 16.8 MB
    _Float16* W1t  = (_Float16*)wsalloc((size_t)EMBED * IN_DIM * 2);   //  1.0 MB
    _Float16* Wkt  = (_Float16*)wsalloc((size_t)EMBED * EMBED * 2);
    _Float16* Wvt  = (_Float16*)wsalloc((size_t)EMBED * EMBED * 2);
    float*    H32  = (float*)   wsalloc((size_t)NROWS * EMBED * 4);    // 16.8 MB
    _Float16* Hh   = (_Float16*)wsalloc((size_t)NROWS * EMBED * 2);    //  8.4 MB
    _Float16* K16  = (_Float16*)wsalloc((size_t)NROWS * EMBED * 2);    //  8.4 MB
    _Float16* V16  = (_Float16*)wsalloc((size_t)NROWS * EMBED * 2);    //  8.4 MB
    float*    q0   = (float*)   wsalloc(EMBED * 4);
    float*    s0   = (float*)   wsalloc(NROWS * 4);
    float*    w0   = (float*)   wsalloc(NROWS * 4);
    float*    at0  = (float*)   wsalloc(EMBED * 4);

    // 1. Convert inputs to f16 (x padded with zero row 0; weights transposed to NxK)
    {
        int total = NROWS * IN_DIM;
        cvt_x_kernel<<<(total + 255) / 256, 256, 0, stream>>>(x, X16, total, IN_DIM);
    }
    transpose_cvt_kernel<<<(IN_DIM * EMBED + 255) / 256, 256, 0, stream>>>(W1, W1t, IN_DIM, EMBED);
    transpose_cvt_kernel<<<(EMBED * EMBED + 255) / 256, 256, 0, stream>>>(Wk, Wkt, EMBED, EMBED);
    transpose_cvt_kernel<<<(EMBED * EMBED + 255) / 256, 256, 0, stream>>>(Wv, Wvt, EMBED, EMBED);

    // 2. Embed GEMM: H = relu(X16 @ W1t^T + b1)  (8192x1024x512), f32 + f16 outputs
    {
        int tilesM = NROWS / 16, tilesN = EMBED / 16;          // 512 x 32
        int blocks = (tilesM * tilesN) / 8;                    // 8 waves per block
        wmma_gemm_kernel<<<blocks, 256, 0, stream>>>(X16, W1t, b1, H32, Hh,
                                                     IN_DIM, EMBED, tilesM, tilesN, 1);
    }

    // 3. Overwrite row 0 with cls_token (must precede K/V GEMMs)
    fix_row0_kernel<<<1, EMBED, 0, stream>>>(cls, H32, Hh);

    // 4. K and V GEMMs: (8192x512x512) each, f16 output
    {
        int tilesM = NROWS / 16, tilesN = EMBED / 16;
        int blocks = (tilesM * tilesN) / 8;
        wmma_gemm_kernel<<<blocks, 256, 0, stream>>>(Hh, Wkt, bk, nullptr, K16,
                                                     EMBED, EMBED, tilesM, tilesN, 0);
        wmma_gemm_kernel<<<blocks, 256, 0, stream>>>(Hh, Wvt, bv, nullptr, V16,
                                                     EMBED, EMBED, tilesM, tilesN, 0);
    }

    // 5. CLS-row path: q0 -> scores -> two-level softmax -> attn0 -> LN + classifier
    q0_kernel<<<1, EMBED, 0, stream>>>(H32, Wq, bq, q0);
    scores_row0_kernel<<<NROWS / 256, 256, 0, stream>>>(q0, K16, s0);
    softmax_row0_kernel<<<1, 512, 0, stream>>>(s0, w0);
    attn0_kernel<<<EMBED / 256, 256, 0, stream>>>(w0, V16, at0);
    final_kernel<<<1, EMBED, 0, stream>>>(H32, at0, Wo, bo, ln_g, ln_b, Wc, bc, out);
}
